// GAT_910533067628
// MI455X (gfx1250) — compile-verified, matrix operand order checked by reading, not run on hardware
//
#include <hip/hip_runtime.h>
#include <hip/hip_bf16.h>

typedef __attribute__((ext_vector_type(16))) __bf16 v16bf;
typedef __attribute__((ext_vector_type(8)))  float  v8f;

union BF16x16 { v16bf v; __bf16 e[16]; };
union F32x8   { v8f  v; float  e[8];  };

#define NEG_SLOPE 0.2f
#define MT_PER_WAVE 8

// ---------------- helpers ----------------
__global__ void k_fill_f32(float* __restrict__ p, float v, int n) {
  int i = blockIdx.x * blockDim.x + threadIdx.x;
  if (i < n) p[i] = v;
}
__global__ void k_fill_u32(unsigned* __restrict__ p, unsigned v, int n) {
  int i = blockIdx.x * blockDim.x + threadIdx.x;
  if (i < n) p[i] = v;
}
__global__ void k_cvt_bf16(const float* __restrict__ s, __bf16* __restrict__ d, int n) {
  int i = blockIdx.x * blockDim.x + threadIdx.x;
  if (i < n) d[i] = (__bf16)s[i];
}

// Convert W[K x Nc] (f32, row-major) into WMMA B-fragment order (bf16):
// Wp[nt][kb][lane][j] so each lane's 16-element fragment is contiguous (32B).
// K == 128 (4 K-blocks). Total elements = K*Nc.
__global__ void k_pack_w(const float* __restrict__ W, __bf16* __restrict__ Wp,
                         int Nc) {
  int i = blockIdx.x * blockDim.x + threadIdx.x;
  if (i >= 128 * Nc) return;
  const int j    = i & 15;          // element within fragment
  const int lane = (i >> 4) & 31;
  const int kb   = (i >> 9) & 3;
  const int nt   = i >> 11;
  const int hl   = lane >> 4;
  const int r    = lane & 15;
  const int v    = j >> 1;
  const int lo   = j & 1;
  const int k    = kb * 32 + ((v < 4) ? 2 * v : 16 + 2 * (v - 4)) + hl * 8 + lo;
  const int col  = nt * 16 + r;
  Wp[i] = (__bf16)W[(size_t)k * Nc + col];
}

// monotone float <-> uint encoding so atomicMax(u32) == float max
__device__ __forceinline__ unsigned f32_ord(float f) {
  unsigned u = __float_as_uint(f);
  return (u & 0x80000000u) ? ~u : (u | 0x80000000u);
}
__device__ __forceinline__ float f32_unord(unsigned u) {
  u = (u & 0x80000000u) ? (u ^ 0x80000000u) : ~u;
  return __uint_as_float(u);
}

// ---------------- WMMA GEMM: C[M,Nc] = A[M,K]@W[K,Nc] + bias (K == 128) -------------
// One 16x16 tile per wave; B fragments (pre-packed) loaded with 4 contiguous 32B
// loads; wave sweeps MT_PER_WAVE M-tiles. Layouts per CDNA5 ISA 7.12.2 (wave32).
template <int NC>
__global__ void k_gemm_bf16(const __bf16* __restrict__ A, const __bf16* __restrict__ Wp,
                            const float* __restrict__ bias, float* __restrict__ C,
                            int M, int K, int Nc_rt) {
  const int Nc   = (NC > 0) ? NC : Nc_rt;
  const int gw   = (int)((blockIdx.x * blockDim.x + threadIdx.x) >> 5);
  const int lane = (int)(threadIdx.x & 31);
  const int hl   = lane >> 4;        // half of wave
  const int r    = lane & 15;
  const int ntiles = Nc >> 4;
  const int nt   = gw % ntiles;
  const int mt0  = (gw / ntiles) * MT_PER_WAVE;
  const int col  = nt * 16 + r;      // this lane's C column

  // Preload packed B fragments for the 4 K-blocks: contiguous 32B per fragment.
  BF16x16 bfr[4];
#pragma unroll
  for (int kb = 0; kb < 4; ++kb)
    bfr[kb].v = *(const v16bf*)(Wp + (((size_t)nt * 4 + kb) * 32 + lane) * 16);

  for (int t = 0; t < MT_PER_WAVE; ++t) {
    const int mt = mt0 + t;
    if (mt * 16 >= M) return;
    int rowA = mt * 16 + r;
    if (rowA >= M) rowA = M - 1;
    const __bf16* __restrict__ arow = A + (size_t)rowA * K;

    v8f acc = {};
#pragma unroll
    for (int kb = 0; kb < 4; ++kb) {
      BF16x16 afr;
#pragma unroll
      for (int v = 0; v < 8; ++v) {
        int k = kb * 32 + ((v < 4) ? 2 * v : 16 + 2 * (v - 4)) + hl * 8;
        afr.e[2 * v]     = arow[k];
        afr.e[2 * v + 1] = arow[k + 1];
      }
      acc = __builtin_amdgcn_wmma_f32_16x16x32_bf16(false, afr.v, false, bfr[kb].v,
                                                    (short)0, acc, false, false);
    }

    const float bv = bias[col];
    F32x8 cf; cf.v = acc;
    float* __restrict__ cb = C + (size_t)(mt * 16 + hl * 8) * Nc + col;
    if (mt * 16 + 16 <= M) {           // full tile: 8 plain stores, immediate offsets
#pragma unroll
      for (int i = 0; i < 8; ++i) cb[(size_t)i * Nc] = cf.e[i] + bv;
    } else {                           // ragged tail
      for (int i = 0; i < 8; ++i) {
        int orow = mt * 16 + hl * 8 + i;
        if (orow < M) cb[(size_t)i * Nc] = cf.e[i] + bv;
      }
    }
  }
}

// ---------------- edge phase ----------------
// one wave per edge: e = att . leaky_relu(xl[src] + xr[dst]); atomicMax into m[dst]
template <int FT>
__global__ void k_edge_logits(const float* __restrict__ xl, const float* __restrict__ xr,
                              const float* __restrict__ att,
                              const int* __restrict__ src, const int* __restrict__ dst,
                              float* __restrict__ eout, unsigned* __restrict__ menc,
                              int E, int F_rt) {
  const int eidx = (int)((blockIdx.x * blockDim.x + threadIdx.x) >> 5);
  const int lane = (int)(threadIdx.x & 31);
  if (eidx >= E) return;
  const int s = src[eidx], d = dst[eidx];
  float part = 0.f;
  if constexpr (FT > 0) {
    constexpr int VEC = FT / 32;
    const float* __restrict__ a = xl + (size_t)s * FT + lane * VEC;
    const float* __restrict__ b = xr + (size_t)d * FT + lane * VEC;
    const float* __restrict__ w = att + lane * VEC;
#pragma unroll
    for (int v = 0; v < VEC; ++v) {
      float z = a[v] + b[v];
      z = z > 0.f ? z : NEG_SLOPE * z;
      part += z * w[v];
    }
  } else {
    const float* __restrict__ a = xl + (size_t)s * F_rt;
    const float* __restrict__ b = xr + (size_t)d * F_rt;
    for (int f = lane; f < F_rt; f += 32) {
      float z = a[f] + b[f];
      z = z > 0.f ? z : NEG_SLOPE * z;
      part += z * att[f];
    }
  }
#pragma unroll
  for (int off = 16; off; off >>= 1) part += __shfl_xor(part, off, 32);
  if (lane == 0) {
    eout[eidx] = part;
    atomicMax(&menc[d], f32_ord(part));
  }
}

// one thread per edge: p = exp(e - m[dst]); s[dst] += p
__global__ void k_edge_exp(const float* __restrict__ e, const unsigned* __restrict__ menc,
                           const int* __restrict__ dst, float* __restrict__ p,
                           float* __restrict__ ssum, int E) {
  int i = blockIdx.x * blockDim.x + threadIdx.x;
  if (i >= E) return;
  const int d = dst[i];
  const float pv = __expf(e[i] - f32_unord(menc[d]));
  p[i] = pv;
  atomicAdd(&ssum[d], pv);
}

// one wave per edge: agg[dst] += (p/s[dst]) * xl[src]
template <int FT>
__global__ void k_edge_scatter(const float* __restrict__ xl, const float* __restrict__ p,
                               const float* __restrict__ ssum,
                               const int* __restrict__ src, const int* __restrict__ dst,
                               float* __restrict__ agg, int E, int F_rt) {
  const int eidx = (int)((blockIdx.x * blockDim.x + threadIdx.x) >> 5);
  const int lane = (int)(threadIdx.x & 31);
  if (eidx >= E) return;
  const int s = src[eidx], d = dst[eidx];
  const float alpha = p[eidx] / ssum[d];
  if constexpr (FT > 0) {
    constexpr int VEC = FT / 32;
    const float* __restrict__ a = xl + (size_t)s * FT + lane * VEC;
    float* __restrict__ o = agg + (size_t)d * FT + lane * VEC;
#pragma unroll
    for (int v = 0; v < VEC; ++v) atomicAdd(&o[v], alpha * a[v]);
  } else {
    const float* __restrict__ a = xl + (size_t)s * F_rt;
    float* __restrict__ o = agg + (size_t)d * F_rt;
    for (int f = lane; f < F_rt; f += 32) atomicAdd(&o[f], alpha * a[f]);
  }
}

// h = relu(agg + bias[f] + skip) -> bf16
__global__ void k_finalize_relu(const float* __restrict__ agg, const float* __restrict__ skip,
                                const float* __restrict__ bias, __bf16* __restrict__ h,
                                int total, int F) {
  int i = blockIdx.x * blockDim.x + threadIdx.x;
  if (i >= total) return;
  float v = agg[i] + bias[i % F] + skip[i];
  h[i] = (__bf16)(v > 0.f ? v : 0.f);
}

// out = agg + bias[f] + skip (fp32)
__global__ void k_finalize_out(const float* __restrict__ agg, const float* __restrict__ skip,
                               const float* __restrict__ bias, float* __restrict__ out,
                               int total, int F) {
  int i = blockIdx.x * blockDim.x + threadIdx.x;
  if (i >= total) return;
  out[i] = agg[i] + bias[i % F] + skip[i];
}

// ---------------- host ----------------
static inline int cdiv(long a, long b) { return (int)((a + b - 1) / b); }

extern "C" void kernel_launch(void* const* d_in, const int* in_sizes, int n_in,
                              void* d_out, int out_size, void* d_ws, size_t ws_size,
                              hipStream_t stream) {
  const float* x     = (const float*)d_in[0];
  const int*   ei    = (const int*)d_in[1];
  const float* Wl1   = (const float*)d_in[2];
  const float* bl1   = (const float*)d_in[3];
  const float* Wr1   = (const float*)d_in[4];
  const float* br1   = (const float*)d_in[5];
  const float* att1  = (const float*)d_in[6];
  const float* bias1 = (const float*)d_in[7];
  const float* Wlin1 = (const float*)d_in[8];
  const float* blin1 = (const float*)d_in[9];
  const float* Wl2   = (const float*)d_in[10];
  const float* bl2   = (const float*)d_in[11];
  const float* Wr2   = (const float*)d_in[12];
  const float* br2   = (const float*)d_in[13];
  const float* att2  = (const float*)d_in[14];
  const float* bias2 = (const float*)d_in[15];
  const float* Wlin2 = (const float*)d_in[16];
  const float* blin2 = (const float*)d_in[17];

  const int H = in_sizes[3];           // 128
  const int D = in_sizes[2] / H;       // 128
  const int O = in_sizes[11];          // 64
  const int N = in_sizes[0] / D;       // 50000
  const int E = in_sizes[1] / 2;       // 1.6M
  const int* src = ei;
  const int* dst = ei + E;

  // workspace carve-up (layer 2 reuses layer-1 activation buffers)
  char* ws = (char*)d_ws;
  size_t off = 0;
  auto take = [&](size_t bytes) -> void* {
    void* q = ws + off;
    off += (bytes + 255) & ~(size_t)255;
    return q;
  };
  __bf16* x_bf   = (__bf16*)take((size_t)N * D * 2);
  __bf16* h_bf   = (__bf16*)take((size_t)N * H * 2);
  __bf16* Wl1b   = (__bf16*)take((size_t)D * H * 2);
  __bf16* Wr1b   = (__bf16*)take((size_t)D * H * 2);
  __bf16* Wlin1b = (__bf16*)take((size_t)D * H * 2);
  __bf16* Wl2b   = (__bf16*)take((size_t)H * O * 2);
  __bf16* Wr2b   = (__bf16*)take((size_t)H * O * 2);
  __bf16* Wlin2b = (__bf16*)take((size_t)H * O * 2);
  float*  xl     = (float*)take((size_t)N * H * 4);
  float*  xr     = (float*)take((size_t)N * H * 4);
  float*  sk     = (float*)take((size_t)N * H * 4);
  float*  agg    = (float*)take((size_t)N * H * 4);
  float*  elog   = (float*)take((size_t)E * 4);
  float*  p      = (float*)take((size_t)E * 4);
  unsigned* menc = (unsigned*)take((size_t)N * 4);
  float*  ssum   = (float*)take((size_t)N * 4);
  (void)ws_size; (void)n_in; (void)out_size;

  const int T = 256;
  const unsigned NEG_INF_ENC = 0x007FFFFFu; // f32_ord(-inf)

  auto gemm = [&](const __bf16* A, const __bf16* Wp, const float* bias, float* C,
                  int M, int K, int Nc) {
    int ntiles  = Nc / 16;
    int wgroups = cdiv(cdiv(M, 16), MT_PER_WAVE);
    int blocks  = cdiv((long)wgroups * ntiles, T / 32);
    if (Nc == 128)      k_gemm_bf16<128><<<blocks, T, 0, stream>>>(A, Wp, bias, C, M, K, Nc);
    else if (Nc == 64)  k_gemm_bf16<64><<<blocks, T, 0, stream>>>(A, Wp, bias, C, M, K, Nc);
    else                k_gemm_bf16<0><<<blocks, T, 0, stream>>>(A, Wp, bias, C, M, K, Nc);
  };
  auto logits = [&](const float* XL, const float* XR, const float* ATT, int F) {
    int blocks = cdiv(E, T / 32);
    if (F == 128)      k_edge_logits<128><<<blocks, T, 0, stream>>>(XL, XR, ATT, src, dst, elog, menc, E, F);
    else if (F == 64)  k_edge_logits<64><<<blocks, T, 0, stream>>>(XL, XR, ATT, src, dst, elog, menc, E, F);
    else               k_edge_logits<0><<<blocks, T, 0, stream>>>(XL, XR, ATT, src, dst, elog, menc, E, F);
  };
  auto scatter = [&](const float* XL, int F) {
    int blocks = cdiv(E, T / 32);
    if (F == 128)      k_edge_scatter<128><<<blocks, T, 0, stream>>>(XL, p, ssum, src, dst, agg, E, F);
    else if (F == 64)  k_edge_scatter<64><<<blocks, T, 0, stream>>>(XL, p, ssum, src, dst, agg, E, F);
    else               k_edge_scatter<0><<<blocks, T, 0, stream>>>(XL, p, ssum, src, dst, agg, E, F);
  };

  // ---- activations -> bf16; weights -> bf16 in WMMA fragment order ----
  k_cvt_bf16<<<cdiv((long)N * D, T), T, 0, stream>>>(x, x_bf, N * D);
  k_pack_w<<<cdiv((long)D * H, T), T, 0, stream>>>(Wl1,   Wl1b,   H);
  k_pack_w<<<cdiv((long)D * H, T), T, 0, stream>>>(Wr1,   Wr1b,   H);
  k_pack_w<<<cdiv((long)D * H, T), T, 0, stream>>>(Wlin1, Wlin1b, H);
  k_pack_w<<<cdiv((long)H * O, T), T, 0, stream>>>(Wl2,   Wl2b,   O);
  k_pack_w<<<cdiv((long)H * O, T), T, 0, stream>>>(Wr2,   Wr2b,   O);
  k_pack_w<<<cdiv((long)H * O, T), T, 0, stream>>>(Wlin2, Wlin2b, O);

  // ---- layer 1 (D -> H) ----
  gemm(x_bf, Wl1b,   bl1,   xl, N, D, H);
  gemm(x_bf, Wr1b,   br1,   xr, N, D, H);
  gemm(x_bf, Wlin1b, blin1, sk, N, D, H);
  k_fill_u32<<<cdiv(N, T), T, 0, stream>>>(menc, NEG_INF_ENC, N);
  k_fill_f32<<<cdiv(N, T), T, 0, stream>>>(ssum, 0.f, N);
  k_fill_f32<<<cdiv((long)N * H, T), T, 0, stream>>>(agg, 0.f, N * H);
  logits(xl, xr, att1, H);
  k_edge_exp<<<cdiv(E, T), T, 0, stream>>>(elog, menc, dst, p, ssum, E);
  scatter(xl, H);
  k_finalize_relu<<<cdiv((long)N * H, T), T, 0, stream>>>(agg, sk, bias1, h_bf, N * H, H);

  // ---- layer 2 (H -> O), reusing xl/xr/sk/agg/elog/p/menc/ssum ----
  gemm(h_bf, Wl2b,   bl2,   xl, N, H, O);
  gemm(h_bf, Wr2b,   br2,   xr, N, H, O);
  gemm(h_bf, Wlin2b, blin2, sk, N, H, O);
  k_fill_u32<<<cdiv(N, T), T, 0, stream>>>(menc, NEG_INF_ENC, N);
  k_fill_f32<<<cdiv(N, T), T, 0, stream>>>(ssum, 0.f, N);
  k_fill_f32<<<cdiv((long)N * O, T), T, 0, stream>>>(agg, 0.f, N * O);
  logits(xl, xr, att2, O);
  k_edge_exp<<<cdiv(E, T), T, 0, stream>>>(elog, menc, dst, p, ssum, E);
  scatter(xl, O);
  k_finalize_out<<<cdiv((long)N * O, T), T, 0, stream>>>(agg, sk, bias2, (float*)d_out, N * O, O);
}
